// Mamba_Block_32590211842131
// MI455X (gfx1250) — matrix operand order repaired and checked by reading, not compiled, not run on hardware
//
#include <hip/hip_runtime.h>
#include <hip/hip_bf16.h>

// Problem constants (match reference)
#define BATCH   4
#define CDIM    256
#define LDIM    2048
#define DSTATE  16
#define DINNER  512
#define DTRANK  16
#define MTOT    (BATCH * LDIM)   // 8192 rows (b*L + l)

typedef __bf16 bf16;
typedef __attribute__((ext_vector_type(16))) __bf16 v16bf;
typedef __attribute__((ext_vector_type(8)))  __bf16 v8bf;
typedef __attribute__((ext_vector_type(8)))  float  v8f;

__device__ __forceinline__ int lane_id() { return (int)(threadIdx.x & 31); }

// ---------------------------------------------------------------------------
// CDNA5 bf16 WMMA fragment loaders (wave32, 16x16x32)
// A-matrix (16x32, row-major [M,K]): lanes 0-15 -> M=lane, K = {0..7, 16..23};
//                                    lanes 16-31 -> M=lane-16, K = {8..15, 24..31}
// ---------------------------------------------------------------------------
__device__ __forceinline__ v16bf load_fragA(const bf16* __restrict__ A,
                                            int row0, int lda, int k0) {
  int lane = lane_id();
  int r  = row0 + (lane & 15);
  int kb = k0 + ((lane & 16) ? 8 : 0);
  const bf16* q = A + (size_t)r * lda + kb;
  v8bf lo = *(const v8bf*)(q);
  v8bf hi = *(const v8bf*)(q + 16);
  v16bf f;
#pragma unroll
  for (int i = 0; i < 8; ++i) { f[i] = lo[i]; f[i + 8] = hi[i]; }
  return f;
}

// B-matrix (32x16): B[k][n] = W[n][k] with W row-major [N,K].
// lanes 0-15 -> N=lane, K = {0..7, 16..23} ; lanes 16-31 -> N=lane-16, K = {8..15, 24..31}
__device__ __forceinline__ v16bf load_fragB(const bf16* __restrict__ W,
                                            int col0, int ldw, int k0) {
  int lane = lane_id();
  int n  = col0 + (lane & 15);
  int kb = k0 + ((lane & 16) ? 8 : 0);
  const bf16* q = W + (size_t)n * ldw + kb;
  v8bf lo = *(const v8bf*)(q);
  v8bf hi = *(const v8bf*)(q + 16);
  v16bf f;
#pragma unroll
  for (int i = 0; i < 8; ++i) { f[i] = lo[i]; f[i + 8] = hi[i]; }
  return f;
}

// ---------------------------------------------------------------------------
// fp32 -> bf16 weight cast (weights stay L2-resident afterwards)
// ---------------------------------------------------------------------------
__global__ void cvt_bf16_kernel(const float* __restrict__ src,
                                bf16* __restrict__ dst, int n) {
  for (int i = blockIdx.x * blockDim.x + threadIdx.x; i < n;
       i += gridDim.x * blockDim.x)
    dst[i] = (bf16)src[i];
}

// ---------------------------------------------------------------------------
// Front LayerNorm over C; input x is (B, C, L) so reads are L-strided.
// One block per (b,l) row; outputs bf16 GEMM operand u (MTOT x CDIM).
// ---------------------------------------------------------------------------
__global__ __launch_bounds__(256) void ln_front_kernel(
    const float* __restrict__ x, const float* __restrict__ lnw,
    const float* __restrict__ lnb, bf16* __restrict__ u) {
  __shared__ float s1[256];
  __shared__ float s2[256];
  int m = blockIdx.x;            // b*L + l
  int b = m >> 11, l = m & (LDIM - 1);
  int c = threadIdx.x;
  float v = x[((size_t)(b * CDIM + c)) * LDIM + l];
  s1[c] = v; s2[c] = v * v;
  __syncthreads();
  for (int s = 128; s > 0; s >>= 1) {
    if (c < s) { s1[c] += s1[c + s]; s2[c] += s2[c + s]; }
    __syncthreads();
  }
  float mu  = s1[0] * (1.0f / CDIM);
  float var = s2[0] * (1.0f / CDIM) - mu * mu;
  float rs  = rsqrtf(var + 1e-5f);
  u[(size_t)m * CDIM + c] = (bf16)((v - mu) * rs * lnw[c] + lnb[c]);
}

// ---------------------------------------------------------------------------
// Register-blocked bf16 WMMA GEMM: C[M,N] = A[M,K] * W[N,K]^T.
// One wave computes a 16 x (16*NT) strip: the A fragment is loaded once per
// K-step and reused by NT WMMAs (NT accumulators), cutting fragment traffic.
// ---------------------------------------------------------------------------
template <int NT>
__global__ __launch_bounds__(32) void wmma_gemm_bf16_nt(
    const bf16* __restrict__ A, const bf16* __restrict__ W,
    float* __restrict__ C, int K, int ldc) {
  int n0 = blockIdx.x * (16 * NT);
  int m0 = blockIdx.y * 16;
  v8f acc[NT];
#pragma unroll
  for (int i = 0; i < NT; ++i) acc[i] = (v8f){0.f,0.f,0.f,0.f,0.f,0.f,0.f,0.f};
  for (int k0 = 0; k0 < K; k0 += 32) {
    v16bf a = load_fragA(A, m0, K, k0);
#pragma unroll
    for (int i = 0; i < NT; ++i) {
      v16bf b = load_fragB(W, n0 + 16 * i, K, k0);
      acc[i] = __builtin_amdgcn_wmma_f32_16x16x32_bf16(false, a, false, b,
                                                       (short)0, acc[i],
                                                       false, false);
    }
  }
  int lane = lane_id();
  int rb   = m0 + ((lane & 16) ? 8 : 0);
#pragma unroll
  for (int i = 0; i < NT; ++i) {
    int col = n0 + 16 * i + (lane & 15);
#pragma unroll
    for (int j = 0; j < 8; ++j)
      C[(size_t)(rb + j) * ldc + col] = acc[i][j];
  }
}

// ---------------------------------------------------------------------------
// Causal depthwise conv (K=4) + bias + SiLU. xz holds [x_in | z] (1024 wide).
// ---------------------------------------------------------------------------
__global__ __launch_bounds__(256) void conv_silu_kernel(
    const float* __restrict__ xz, const float* __restrict__ cw,
    const float* __restrict__ cb, float* __restrict__ xc,
    bf16* __restrict__ xcb) {
  int idx = blockIdx.x * 256 + threadIdx.x;   // m*512 + d
  int m = idx >> 9, d = idx & (DINNER - 1);
  int b = m >> 11, l = m & (LDIM - 1);
  float acc = cb[d];
#pragma unroll
  for (int j = 0; j < 4; ++j) {
    int ls = l - 3 + j;
    if (ls >= 0)
      acc += cw[d * 4 + j] * xz[((size_t)(b * LDIM + ls)) * 1024 + d];
  }
  float s = acc / (1.0f + expf(-acc));
  xc[idx]  = s;
  xcb[idx] = (bf16)s;
}

// ---------------------------------------------------------------------------
// delta = softplus(dt @ dt_proj_w^T + bias), K = DT_RANK = 16 (VALU dot)
// ---------------------------------------------------------------------------
__global__ __launch_bounds__(256) void delta_kernel(
    const float* __restrict__ xdbl, const float* __restrict__ dtw,
    const float* __restrict__ dtb, float* __restrict__ dlt) {
  int idx = blockIdx.x * 256 + threadIdx.x;   // m*512 + d
  int m = idx >> 9, d = idx & (DINNER - 1);
  float acc = dtb[d];
  const float* r = xdbl + (size_t)m * 48;
#pragma unroll
  for (int j = 0; j < DTRANK; ++j) acc += r[j] * dtw[d * DTRANK + j];
  dlt[idx] = (acc > 20.0f) ? acc : log1pf(expf(acc));
}

// ---------------------------------------------------------------------------
// Selective scan: 2048 (b,d) channels, 2 channels per wave32,
// D_STATE=16 mapped to 16 lanes; h.C reduced with 4 shfl_xor hops.
// Fused D-skip and SiLU(z) gating; emits bf16 operand for out_proj.
// ---------------------------------------------------------------------------
__global__ __launch_bounds__(128) void scan_kernel(
    const float* __restrict__ dlt, const float* __restrict__ xc,
    const float* __restrict__ xdbl, const float* __restrict__ xz,
    const float* __restrict__ Alog, const float* __restrict__ Dp,
    bf16* __restrict__ ybf) {
  int lane = lane_id();
  int wave = (blockIdx.x * blockDim.x + threadIdx.x) >> 5;
  int ch = wave * 2 + ((lane & 16) ? 1 : 0);      // channel = b*512 + d
  int n = lane & 15;
  int b = ch >> 9, d = ch & (DINNER - 1);
  float a  = -expf(Alog[d * DSTATE + n]);
  float Dd = Dp[d];
  float h = 0.0f;
  size_t mbase = (size_t)b * LDIM;
  for (int t = 0; t < LDIM; ++t) {
    size_t m = mbase + t;
    float dl = dlt[m * DINNER + d];
    float xv = xc[m * DINNER + d];
    const float* row = xdbl + m * 48;
    float Bv = row[DTRANK + n];
    float Cv = row[DTRANK + DSTATE + n];
    h = expf(dl * a) * h + dl * Bv * xv;
    float p = h * Cv;
    p += __shfl_xor(p, 1, 32);
    p += __shfl_xor(p, 2, 32);
    p += __shfl_xor(p, 4, 32);
    p += __shfl_xor(p, 8, 32);
    if (n == 0) {
      float zv = xz[m * 1024 + DINNER + d];
      float y  = (p + Dd * xv) * (zv / (1.0f + expf(-zv)));
      ybf[m * DINNER + d] = (bf16)y;
    }
  }
}

// ---------------------------------------------------------------------------
// Fused out_proj WMMA + residual + LayerNorm + transposed store.
// Block = 4 waves; each wave owns a 16x64 strip (4 N-tiles, shared A frag),
// results staged in LDS, then each wave LayerNorms 4 of the 16 rows.
// ---------------------------------------------------------------------------
__global__ __launch_bounds__(128) void out_fused_kernel(
    const bf16* __restrict__ Y, const bf16* __restrict__ Wo,
    const float* __restrict__ x, const float* __restrict__ lnw,
    const float* __restrict__ lnb, float* __restrict__ out) {
  __shared__ float tile[16][264];   // padded against bank conflicts
  int m0   = blockIdx.x * 16;
  int wid  = threadIdx.x >> 5;      // 0..3
  int lane = threadIdx.x & 31;
  int n0   = wid * 64;
  v8f acc[4];
#pragma unroll
  for (int i = 0; i < 4; ++i) acc[i] = (v8f){0.f,0.f,0.f,0.f,0.f,0.f,0.f,0.f};
  for (int k0 = 0; k0 < DINNER; k0 += 32) {
    v16bf a = load_fragA(Y, m0, DINNER, k0);
#pragma unroll
    for (int i = 0; i < 4; ++i) {
      v16bf b = load_fragB(Wo, n0 + 16 * i, DINNER, k0);
      acc[i] = __builtin_amdgcn_wmma_f32_16x16x32_bf16(false, a, false, b,
                                                       (short)0, acc[i],
                                                       false, false);
    }
  }
  int rb = (lane & 16) ? 8 : 0;
#pragma unroll
  for (int i = 0; i < 4; ++i) {
    int col = n0 + 16 * i + (lane & 15);
#pragma unroll
    for (int j = 0; j < 8; ++j) tile[rb + j][col] = acc[i][j];
  }
  __syncthreads();

  // LN + residual stage: wave `wid` handles rows 4*wid .. 4*wid+3
#pragma unroll
  for (int rr = 0; rr < 4; ++rr) {
    int r = wid * 4 + rr;
    int m = m0 + r;
    int b = m >> 11, l = m & (LDIM - 1);
    float v[8];
    float sum = 0.f, sq = 0.f;
#pragma unroll
    for (int i = 0; i < 8; ++i) {
      int c = lane + 32 * i;
      float t = tile[r][c] + x[((size_t)(b * CDIM + c)) * LDIM + l];
      v[i] = t; sum += t; sq += t * t;
    }
    for (int mk = 16; mk >= 1; mk >>= 1) {
      sum += __shfl_xor(sum, mk, 32);
      sq  += __shfl_xor(sq,  mk, 32);
    }
    float mu = sum * (1.0f / CDIM);
    float rs = rsqrtf(sq * (1.0f / CDIM) - mu * mu + 1e-5f);
#pragma unroll
    for (int i = 0; i < 8; ++i) {
      int c = lane + 32 * i;
      out[((size_t)(b * CDIM + c)) * LDIM + l] = (v[i] - mu) * rs * lnw[c] + lnb[c];
    }
  }
}

// ---------------------------------------------------------------------------
extern "C" void kernel_launch(void* const* d_in, const int* in_sizes, int n_in,
                              void* d_out, int out_size, void* d_ws, size_t ws_size,
                              hipStream_t stream) {
  (void)in_sizes; (void)n_in; (void)out_size; (void)ws_size;
  const float* x     = (const float*)d_in[0];
  const float* lnw   = (const float*)d_in[1];
  const float* lnb   = (const float*)d_in[2];
  const float* w_in  = (const float*)d_in[3];   // (1024, 256)
  const float* cw    = (const float*)d_in[4];   // (512, 1, 4)
  const float* cb    = (const float*)d_in[5];
  const float* w_xp  = (const float*)d_in[6];   // (48, 512)
  const float* dtw   = (const float*)d_in[7];   // (512, 16)
  const float* dtb   = (const float*)d_in[8];
  const float* Alog  = (const float*)d_in[9];   // (512, 16)
  const float* Dp    = (const float*)d_in[10];
  const float* w_out = (const float*)d_in[11];  // (256, 512)
  float* out = (float*)d_out;

  uint8_t* base = (uint8_t*)d_ws;
  size_t off = 0;
  auto alloc = [&](size_t bytes) {
    uint8_t* p = base + off;
    off = (off + bytes + 255) & ~(size_t)255;
    return p;
  };
  bf16*  u_b  = (bf16*) alloc((size_t)MTOT * CDIM * sizeof(bf16));     // 4 MB
  float* xz   = (float*)alloc((size_t)MTOT * 1024 * sizeof(float));    // 33.5 MB
  float* xc   = (float*)alloc((size_t)MTOT * DINNER * sizeof(float));  // 16.8 MB
  bf16*  xc_b = (bf16*) alloc((size_t)MTOT * DINNER * sizeof(bf16));   // 8.4 MB
  float* xdbl = (float*)alloc((size_t)MTOT * 48 * sizeof(float));      // 1.6 MB
  float* dlt  = (float*)alloc((size_t)MTOT * DINNER * sizeof(float));  // 16.8 MB
  bf16*  y_b  = (bf16*) alloc((size_t)MTOT * DINNER * sizeof(bf16));   // 8.4 MB
  bf16*  wi_b = (bf16*) alloc((size_t)1024 * 256 * sizeof(bf16));
  bf16*  wx_b = (bf16*) alloc((size_t)48 * 512 * sizeof(bf16));
  bf16*  wo_b = (bf16*) alloc((size_t)256 * 512 * sizeof(bf16));

  // Weight casts (stay hot in the 192 MB L2 for the whole pipeline)
  cvt_bf16_kernel<<<512, 256, 0, stream>>>(w_in, wi_b, 1024 * 256);
  cvt_bf16_kernel<<<96, 256, 0, stream>>>(w_xp, wx_b, 48 * 512);
  cvt_bf16_kernel<<<512, 256, 0, stream>>>(w_out, wo_b, 256 * 512);

  // 1) front LayerNorm -> bf16 u
  ln_front_kernel<<<MTOT, 256, 0, stream>>>(x, lnw, lnb, u_b);

  // 2) in_proj GEMM: (8192x256) x (256->1024) -> xz  (16x64 strip per wave)
  wmma_gemm_bf16_nt<4><<<dim3(1024 / 64, MTOT / 16), 32, 0, stream>>>(
      u_b, wi_b, xz, CDIM, 1024);

  // 3) depthwise causal conv + SiLU -> xc (f32 + bf16)
  conv_silu_kernel<<<(MTOT * DINNER) / 256, 256, 0, stream>>>(xz, cw, cb, xc, xc_b);

  // 4) x_proj GEMM: (8192x512) x (512->48) -> x_dbl [dt|B|C] (16x48 per wave)
  wmma_gemm_bf16_nt<3><<<dim3(1, MTOT / 16), 32, 0, stream>>>(
      xc_b, wx_b, xdbl, DINNER, 48);

  // 5) dt_proj + softplus -> delta
  delta_kernel<<<(MTOT * DINNER) / 256, 256, 0, stream>>>(xdbl, dtw, dtb, dlt);

  // 6) selective scan fused with D-skip + SiLU(z) gating -> y (bf16)
  scan_kernel<<<(BATCH * DINNER) / 8, 128, 0, stream>>>(
      dlt, xc, xdbl, xz, Alog, Dp, y_b);

  // 7) out_proj WMMA + residual + LayerNorm + transposed store
  out_fused_kernel<<<MTOT / 16, 128, 0, stream>>>(y_b, wo_b, x, lnw, lnb, out);
}
